// TurboQuantMSE_37933151158999
// MI455X (gfx1250) — matrix operand order repaired and checked by reading, not compiled, not run on hardware
//
#include <hip/hip_runtime.h>

// TurboQuant MSE on MI455X (gfx1250):
//   y = x @ Pi^T  (bf16 WMMA)  -> 4-bit Lloyd-Max quantize -> y_hat (bf16, ws)
//   x_hat = y_hat @ Pi (bf16 WMMA) -> f32 out
// Outputs: d_out = [ x_hat (4096*4096 f32) | indices (4096*4096, stored as f32) ]
//
// Fast path (ws >= 128MB): pre-convert x/Pi/Pi^T to bf16, then both GEMMs are
// NT-layout with double-buffered GLOBAL_LOAD_ASYNC_TO_LDS_B128 staging
// (ASYNCcnt-tracked DMA overlapping the WMMA pipeline).
// Fallback path (small ws): synchronous fp32->bf16 staging (round-1 kernels).

#define DIM      4096
#define BM       128
#define BN       128
#define BK       32
#define APITCH   40       // 32 data halves + 8 pad halves per LDS row (80B = 5*16B)
#define NTHREADS 256

typedef __attribute__((ext_vector_type(16))) __bf16 bf16x16;
typedef __attribute__((ext_vector_type(8)))  float  floatx8;

union Frag {
    bf16x16 v;
    uint4   q[2];
};

__device__ __forceinline__ unsigned short f2bf(float f) {
    union { __bf16 h; unsigned short u; } cv;
    cv.h = (__bf16)f;
    return cv.u;
}

// ---------------------------------------------------------------------------
// Shared fragment loaders (ISA 7.12.2 layouts)
// A fragment (16x32 bf16): lane L holds row M=L%16; h=L/16 selects K-chunks
// {h*8..h*8+7} and {16+h*8..16+h*8+7}.
__device__ __forceinline__ void load_a(Frag& a, const unsigned short* As, int mrow, int r, int h) {
    const unsigned short* p = As + (mrow + r) * APITCH;
    a.q[0] = *(const uint4*)(p + h * 8);
    a.q[1] = *(const uint4*)(p + 16 + h * 8);
}
// B fragment (32x16 bf16): lane L holds column N=L%16; K-chunks {h*8..h*8+7} and {16+h*8..16+h*8+7}.
__device__ __forceinline__ void load_b(Frag& b, const unsigned short* Bs, int ncol, int r, int h) {
    const unsigned short* p = Bs + (ncol + r) * APITCH;
    b.q[0] = *(const uint4*)(p + h * 8);
    b.q[1] = *(const uint4*)(p + 16 + h * 8);
}

// ---------------------------------------------------------------------------
// FAST PATH: bf16 operands in ws, async DMA staging, double buffered.
// ---------------------------------------------------------------------------

// Issue async copy of one 128x32 bf16 tile (row pitch DIM halves) into LDS.
// Per thread: 32 bytes = 2 x b128. INST_OFFSET applies to BOTH the LDS and the
// global address (ISA 15.18.3), so one (ldsOff, vOff) pair serves both halves.
__device__ __forceinline__ void async_stage_tile(unsigned short* ldsTile,
                                                 const unsigned short* gtile /* row0,k0 origin */) {
    const int t   = threadIdx.x;
    const int row = t >> 1;
    const int seg = (t & 1) * 16;                       // halves
    unsigned ldsOff = (unsigned)(uintptr_t)(ldsTile + row * APITCH + seg);
    unsigned vOff   = (unsigned)((row * DIM + seg) * 2); // byte offset from gtile
    asm volatile("global_load_async_to_lds_b128 %0, %1, %2 offset:0\n\t"
                 "global_load_async_to_lds_b128 %0, %1, %2 offset:16"
                 :: "v"(ldsOff), "v"(vOff), "s"(gtile)
                 : "memory");
}

__device__ __forceinline__ void wait_async_le4() {
    asm volatile("s_wait_asynccnt 0x4" ::: "memory");
}
__device__ __forceinline__ void wait_async_0() {
    asm volatile("s_wait_asynccnt 0x0" ::: "memory");
}

// One NT GEMM: C[row,col] = sum_k A[row,k]*B[col,k], A/B bf16 row-major 4096x4096.
// QUANT=true : fused 4-bit quantize -> outF=indices(as float), yhat=centroid values (bf16)
// QUANT=false: outF = C (f32)
template <bool QUANT>
__global__ __launch_bounds__(NTHREADS)
void tq_gemm_async(const unsigned short* __restrict__ Abase,
                   const unsigned short* __restrict__ Bbase,
                   const float* __restrict__ cent,
                   float* __restrict__ outF,
                   unsigned short* __restrict__ yhat) {
    __shared__ __align__(16) unsigned short As[2][BM * APITCH];
    __shared__ __align__(16) unsigned short Bs[2][BN * APITCH];
    __shared__ float cLds[16];

    const int t = threadIdx.x;

    float bnd[15];
    if constexpr (QUANT) {
        if (t < 16) cLds[t] = cent[t];
#pragma unroll
        for (int i = 0; i < 15; i++) bnd[i] = 0.5f * (cent[i] + cent[i + 1]);
    }

    const int rowBase = blockIdx.y * BM;
    const int colBase = blockIdx.x * BN;

    const int lane  = t & 31;
    const int wave  = t >> 5;
    const int r     = lane & 15;
    const int h     = lane >> 4;
    const int waveM = (wave >> 1) * 32;
    const int waveN = (wave & 1) * 64;

    const unsigned short* Arow = Abase + (size_t)rowBase * DIM;
    const unsigned short* Brow = Bbase + (size_t)colBase * DIM;

    floatx8 acc[2][4];
#pragma unroll
    for (int mi = 0; mi < 2; mi++)
#pragma unroll
        for (int ni = 0; ni < 4; ni++)
            acc[mi][ni] = (floatx8){0.f, 0.f, 0.f, 0.f, 0.f, 0.f, 0.f, 0.f};

    // Preload tile 0 into buffer 0.
    async_stage_tile(As[0], Arow);
    async_stage_tile(Bs[0], Brow);

    const int NT = DIM / BK;
    for (int it = 0; it < NT; ++it) {
        const int cur = it & 1;
        if (it + 1 < NT) {
            // Stream next tile into the other buffer while we compute on `cur`.
            async_stage_tile(As[1 - cur], Arow + (it + 1) * BK);
            async_stage_tile(Bs[1 - cur], Brow + (it + 1) * BK);
            wait_async_le4();   // in-order completion: oldest 4 (tile `it`) landed
        } else {
            wait_async_0();
        }
        __syncthreads();        // all waves' tile `it` data visible

        Frag a[2], b[4];
#pragma unroll
        for (int mi = 0; mi < 2; mi++) load_a(a[mi], As[cur], waveM + mi * 16, r, h);
#pragma unroll
        for (int ni = 0; ni < 4; ni++) load_b(b[ni], Bs[cur], waveN + ni * 16, r, h);

#pragma unroll
        for (int mi = 0; mi < 2; mi++)
#pragma unroll
            for (int ni = 0; ni < 4; ni++)
                acc[mi][ni] = __builtin_amdgcn_wmma_f32_16x16x32_bf16(
                    false, a[mi].v, false, b[ni].v, (short)0, acc[mi][ni], false, false);

        __syncthreads();        // done reading buf `cur` before it is overwritten
    }

#pragma unroll
    for (int mi = 0; mi < 2; mi++) {
#pragma unroll
        for (int ni = 0; ni < 4; ni++) {
#pragma unroll
            for (int j = 0; j < 8; j++) {
                int row = rowBase + waveM + mi * 16 + h * 8 + j;
                int col = colBase + waveN + ni * 16 + r;
                size_t off = (size_t)row * DIM + col;
                float y = acc[mi][ni][j];
                if constexpr (QUANT) {
                    int idx = 0;
#pragma unroll
                    for (int q = 0; q < 15; q++) idx += (y > bnd[q]) ? 1 : 0;
                    outF[off] = (float)idx;
                    yhat[off] = f2bf(cLds[idx]);
                } else {
                    outF[off] = y;
                }
            }
        }
    }
}

// fp32 -> bf16 bulk convert (vectorized, grid-stride over float4 groups).
__global__ __launch_bounds__(256)
void k_cvt_bf16(const float* __restrict__ src, unsigned short* __restrict__ dst, int n4) {
    int i = blockIdx.x * blockDim.x + threadIdx.x;
    int stride = gridDim.x * blockDim.x;
    for (; i < n4; i += stride) {
        float4 f = ((const float4*)src)[i];
        ushort4 hv;
        hv.x = f2bf(f.x); hv.y = f2bf(f.y); hv.z = f2bf(f.z); hv.w = f2bf(f.w);
        ((ushort4*)dst)[i] = hv;
    }
}

// PiT[d][k] = bf16(Pi[k][d]) — LDS-tiled 32x32 transpose, coalesced both sides.
__global__ __launch_bounds__(256)
void k_transpose_bf16(const float* __restrict__ Pi, unsigned short* __restrict__ PiT) {
    __shared__ unsigned short tile[32][34];
    const int bd = blockIdx.x * 32;   // d
    const int bk = blockIdx.y * 32;   // k
    const int tx = threadIdx.x & 31;
    const int ty = threadIdx.x >> 5;  // 0..7
#pragma unroll
    for (int j = 0; j < 4; j++) {
        int k = ty + j * 8;
        tile[k][tx] = f2bf(Pi[(size_t)(bk + k) * DIM + bd + tx]);
    }
    __syncthreads();
#pragma unroll
    for (int j = 0; j < 4; j++) {
        int d = ty + j * 8;
        PiT[(size_t)(bd + d) * DIM + bk + tx] = tile[tx][d];
    }
}

// ---------------------------------------------------------------------------
// FALLBACK PATH (small ws): synchronous fp32->bf16 staging.
// ---------------------------------------------------------------------------

__device__ __forceinline__ void stage_f32_tile(unsigned short* dst, const float* src) {
    int t   = threadIdx.x;
    int row = t >> 1;
    int seg = (t & 1) * 16;
    const float*    s = src + (size_t)row * DIM + seg;
    unsigned short* d = dst + row * APITCH + seg;
    __builtin_prefetch(s + BK, 0, 0);
#pragma unroll
    for (int i = 0; i < 16; i += 4) {
        float4 f = *(const float4*)(s + i);
        ushort4 hv;
        hv.x = f2bf(f.x); hv.y = f2bf(f.y); hv.z = f2bf(f.z); hv.w = f2bf(f.w);
        *(ushort4*)(d + i) = hv;
    }
}

__device__ __forceinline__ void stage_bf16_tile(unsigned short* dst, const unsigned short* src) {
    int t   = threadIdx.x;
    int row = t >> 1;
    int seg = (t & 1) * 16;
    const uint4* s = (const uint4*)(src + (size_t)row * DIM + seg);
    uint4*       d = (uint4*)(dst + row * APITCH + seg);
    __builtin_prefetch((const void*)(src + (size_t)row * DIM + seg + BK), 0, 0);
    d[0] = s[0];
    d[1] = s[1];
}

__device__ __forceinline__ void stage_f32_tile_T(unsigned short* dst, const float* src) {
    int t     = threadIdx.x;
    int dcol  = t & 127;
    int kbase = t >> 7;
#pragma unroll
    for (int i = 0; i < 16; i++) {
        int k  = kbase + i * 2;
        float f = src[(size_t)k * DIM + dcol];
        dst[dcol * APITCH + k] = f2bf(f);
    }
}

__global__ __launch_bounds__(NTHREADS)
void tq_gemm1_quant(const float* __restrict__ X, const float* __restrict__ Pi,
                    const float* __restrict__ cent,
                    float* __restrict__ outIdx, unsigned short* __restrict__ yhat) {
    __shared__ __align__(16) unsigned short As[BM * APITCH];
    __shared__ __align__(16) unsigned short Bs[BN * APITCH];
    __shared__ float cLds[16];

    int t = threadIdx.x;
    if (t < 16) cLds[t] = cent[t];

    float bnd[15];
#pragma unroll
    for (int i = 0; i < 15; i++) bnd[i] = 0.5f * (cent[i] + cent[i + 1]);

    const int rowBase = blockIdx.y * BM;
    const int colBase = blockIdx.x * BN;
    const int lane  = t & 31;
    const int wave  = t >> 5;
    const int r     = lane & 15;
    const int h     = lane >> 4;
    const int waveM = (wave >> 1) * 32;
    const int waveN = (wave & 1) * 64;

    floatx8 acc[2][4];
#pragma unroll
    for (int mi = 0; mi < 2; mi++)
#pragma unroll
        for (int ni = 0; ni < 4; ni++)
            acc[mi][ni] = (floatx8){0.f, 0.f, 0.f, 0.f, 0.f, 0.f, 0.f, 0.f};

    for (int k0 = 0; k0 < DIM; k0 += BK) {
        __syncthreads();
        stage_f32_tile(As, X  + (size_t)rowBase * DIM + k0);
        stage_f32_tile(Bs, Pi + (size_t)colBase * DIM + k0);
        __syncthreads();

        Frag a[2], b[4];
#pragma unroll
        for (int mi = 0; mi < 2; mi++) load_a(a[mi], As, waveM + mi * 16, r, h);
#pragma unroll
        for (int ni = 0; ni < 4; ni++) load_b(b[ni], Bs, waveN + ni * 16, r, h);

#pragma unroll
        for (int mi = 0; mi < 2; mi++)
#pragma unroll
            for (int ni = 0; ni < 4; ni++)
                acc[mi][ni] = __builtin_amdgcn_wmma_f32_16x16x32_bf16(
                    false, a[mi].v, false, b[ni].v, (short)0, acc[mi][ni], false, false);
    }

#pragma unroll
    for (int mi = 0; mi < 2; mi++)
#pragma unroll
        for (int ni = 0; ni < 4; ni++)
#pragma unroll
            for (int j = 0; j < 8; j++) {
                int row = rowBase + waveM + mi * 16 + h * 8 + j;
                int col = colBase + waveN + ni * 16 + r;
                float y = acc[mi][ni][j];
                int idx = 0;
#pragma unroll
                for (int q = 0; q < 15; q++) idx += (y > bnd[q]) ? 1 : 0;
                size_t off = (size_t)row * DIM + col;
                outIdx[off] = (float)idx;
                yhat[off]   = f2bf(cLds[idx]);
            }
}

__global__ __launch_bounds__(NTHREADS)
void tq_gemm2(const unsigned short* __restrict__ yhat, const float* __restrict__ Pi,
              float* __restrict__ xhat) {
    __shared__ __align__(16) unsigned short As[BM * APITCH];
    __shared__ __align__(16) unsigned short BsT[BN * APITCH];

    const int rowBase = blockIdx.y * BM;
    const int colBase = blockIdx.x * BN;
    const int t     = threadIdx.x;
    const int lane  = t & 31;
    const int wave  = t >> 5;
    const int r     = lane & 15;
    const int h     = lane >> 4;
    const int waveM = (wave >> 1) * 32;
    const int waveN = (wave & 1) * 64;

    floatx8 acc[2][4];
#pragma unroll
    for (int mi = 0; mi < 2; mi++)
#pragma unroll
        for (int ni = 0; ni < 4; ni++)
            acc[mi][ni] = (floatx8){0.f, 0.f, 0.f, 0.f, 0.f, 0.f, 0.f, 0.f};

    for (int k0 = 0; k0 < DIM; k0 += BK) {
        __syncthreads();
        stage_bf16_tile(As, yhat + (size_t)rowBase * DIM + k0);
        stage_f32_tile_T(BsT, Pi + (size_t)k0 * DIM + colBase);
        __syncthreads();

        Frag a[2], b[4];
#pragma unroll
        for (int mi = 0; mi < 2; mi++) load_a(a[mi], As, waveM + mi * 16, r, h);
#pragma unroll
        for (int ni = 0; ni < 4; ni++) load_b(b[ni], BsT, waveN + ni * 16, r, h);

#pragma unroll
        for (int mi = 0; mi < 2; mi++)
#pragma unroll
            for (int ni = 0; ni < 4; ni++)
                acc[mi][ni] = __builtin_amdgcn_wmma_f32_16x16x32_bf16(
                    false, a[mi].v, false, b[ni].v, (short)0, acc[mi][ni], false, false);
    }

#pragma unroll
    for (int mi = 0; mi < 2; mi++)
#pragma unroll
        for (int ni = 0; ni < 4; ni++)
#pragma unroll
            for (int j = 0; j < 8; j++) {
                int row = rowBase + waveM + mi * 16 + h * 8 + j;
                int col = colBase + waveN + ni * 16 + r;
                xhat[(size_t)row * DIM + col] = acc[mi][ni][j];
            }
}

// ---------------------------------------------------------------------------

extern "C" void kernel_launch(void* const* d_in, const int* in_sizes, int n_in,
                              void* d_out, int out_size, void* d_ws, size_t ws_size,
                              hipStream_t stream) {
    const float* x    = (const float*)d_in[0];
    const float* Pi   = (const float*)d_in[1];
    const float* cent = (const float*)d_in[2];

    float* xhat   = (float*)d_out;
    float* outIdx = (float*)d_out + (size_t)DIM * DIM;

    const size_t HALF_MAT = (size_t)DIM * DIM * sizeof(unsigned short); // 32 MB

    dim3 grid(DIM / BN, DIM / BM);
    dim3 block(NTHREADS);

    if (ws_size >= 4 * HALF_MAT) {
        // Fast path: bf16 operands + async-DMA double-buffered GEMMs.
        unsigned short* Xb   = (unsigned short*)d_ws;
        unsigned short* Pib  = Xb  + (size_t)DIM * DIM;
        unsigned short* PiT  = Pib + (size_t)DIM * DIM;
        unsigned short* yhat = PiT + (size_t)DIM * DIM;

        const int n4 = (DIM * DIM) / 4;
        hipLaunchKernelGGL(k_cvt_bf16, dim3(2048), dim3(256), 0, stream, x,  Xb,  n4);
        hipLaunchKernelGGL(k_cvt_bf16, dim3(2048), dim3(256), 0, stream, Pi, Pib, n4);
        hipLaunchKernelGGL(k_transpose_bf16, dim3(DIM / 32, DIM / 32), dim3(256), 0, stream, Pi, PiT);

        hipLaunchKernelGGL((tq_gemm_async<true>),  grid, block, 0, stream, Xb,   Pib, cent, outIdx, yhat);
        hipLaunchKernelGGL((tq_gemm_async<false>), grid, block, 0, stream, yhat, PiT, cent, xhat,   (unsigned short*)nullptr);
    } else {
        // Fallback: needs only 32 MB ws for y_hat.
        unsigned short* yhat = (unsigned short*)d_ws;
        hipLaunchKernelGGL(tq_gemm1_quant, grid, block, 0, stream, x, Pi, cent, outIdx, yhat);
        hipLaunchKernelGGL(tq_gemm2,       grid, block, 0, stream, yhat, Pi, xhat);
    }
}
